// BottleneckAttention_68719477249
// MI455X (gfx1250) — matrix operand order repaired and checked by reading, not compiled, not run on hardware
//
#include <hip/hip_runtime.h>

// ---------------------------------------------------------------------------
// BottleneckAttention on MI455X (gfx1250): flash-fused relative-pos attention.
// GEMMs via v_wmma_f32_16x16x32_f16 (wave32, fp32 accum); K/V tiles staged to
// LDS with double-buffered global_load_async_to_lds_b128 (ASYNCcnt path).
// ---------------------------------------------------------------------------

#define HEADS 4
#define BDIM  2
#define CDIM  256
#define HDIM  64
#define WDIM  64
#define NPIX  4096          // HDIM*WDIM
#define DH    64            // CDIM/HEADS
#define QSCALE 0.125f       // DH^-0.5

typedef __attribute__((ext_vector_type(16))) _Float16 v16h;
typedef __attribute__((ext_vector_type(8)))  _Float16 v8h;
typedef __attribute__((ext_vector_type(8)))  float    v8f;

static __device__ __forceinline__ v8f wmma_f16(v16h a, v16h b, v8f c) {
  // (neg_a, A, neg_b, B, c_mod, C, reuse_a, reuse_b)
  return __builtin_amdgcn_wmma_f32_16x16x32_f16(false, a, false, b, (short)0, c,
                                                false, false);
}

// A-matrix 16x32 f16 fragment from a row-major source (row stride `ld` halves).
// Lane L holds row L%16; K runs {b8..b8+7} and {16+b8..16+b8+7}, b8 = 8*(L>=16)
// (per CDNA5 ISA 16-bit A layout) -> two aligned 16B loads.
static __device__ __forceinline__ v16h load_a(const _Float16* base, int ld, int koff) {
  int lane = threadIdx.x & 31;
  int r    = lane & 15;
  int b8   = (lane & 16) ? 8 : 0;
  const _Float16* p = base + (size_t)r * ld + koff + b8;
  v8h lo = *(const v8h*)p;
  v8h hi = *(const v8h*)(p + 16);
  return __builtin_shufflevector(lo, hi, 0, 1, 2, 3, 4, 5, 6, 7,
                                 8, 9, 10, 11, 12, 13, 14, 15);
}

// B-matrix 32x16 f16 fragment: column n of B = row n of the (row-major) source.
// Lane L holds column L%16; K runs {b8..b8+7} and {16+b8..16+b8+7}, b8 = 8*(L>=16).
static __device__ __forceinline__ v16h load_b(const _Float16* base, int ld, int koff) {
  int lane = threadIdx.x & 31;
  int n    = lane & 15;
  int b8   = (lane & 16) ? 8 : 0;
  const _Float16* p = base + (size_t)n * ld + koff + b8;
  v8h lo = *(const v8h*)p;
  v8h hi = *(const v8h*)(p + 16);
  return __builtin_shufflevector(lo, hi, 0, 1, 2, 3, 4, 5, 6, 7,
                                 8, 9, 10, 11, 12, 13, 14, 15);
}

// --------------------------- conversion kernels ----------------------------

// x [B,C,N] f32 -> xt [B,N,C] f16 (so GEMM B-columns are contiguous)
__global__ void cvt_xt_kernel(const float* __restrict__ x, _Float16* __restrict__ xt) {
  int i = blockIdx.x * blockDim.x + threadIdx.x;
  if (i >= BDIM * CDIM * NPIX) return;
  int n  = i & (NPIX - 1);
  int ch = (i >> 12) & (CDIM - 1);
  int b  = i >> 20;
  xt[((size_t)b * NPIX + n) * CDIM + ch] = (_Float16)x[i];
}

__global__ void cvt_f16_kernel(const float* __restrict__ src,
                               _Float16* __restrict__ dst, int count) {
  int i = blockIdx.x * blockDim.x + threadIdx.x;
  if (i < count) dst[i] = (_Float16)src[i];
}

// ------------------------------ QKV projection -----------------------------
// qkv[o,n] = sum_c w_qkv[o,c] * x[c,n]; scatter into head-major Q/K and V^T.
// grid (NPIX/64, 3C/16, B), block 128 (4 waves, one 16x16 tile each).
__global__ void qkv_kernel(const _Float16* __restrict__ wq,   // [3C,C]
                           const _Float16* __restrict__ xt,   // [B,N,C]
                           _Float16* __restrict__ qh,         // [B,H,N,DH] (scaled)
                           _Float16* __restrict__ kh,         // [B,H,N,DH]
                           _Float16* __restrict__ vth) {      // [B,H,DH,N]
  int wave = threadIdx.x >> 5;
  int lane = threadIdx.x & 31;
  int n0 = (blockIdx.x * 4 + wave) * 16;
  int o0 = blockIdx.y * 16;
  int batch = blockIdx.z;
  const _Float16* a0 = wq + (size_t)o0 * CDIM;
  const _Float16* b0 = xt + ((size_t)batch * NPIX + n0) * CDIM;
  v8f acc = {};
#pragma unroll
  for (int kk = 0; kk < CDIM; kk += 32)
    acc = wmma_f16(load_a(a0, CDIM, kk), load_b(b0, CDIM, kk), acc);

  int col   = lane & 15;
  int half8 = (lane & 16) ? 8 : 0;
  int n   = n0 + col;
  int sec = o0 >> 8;  // 0=q 1=k 2=v, uniform per block
#pragma unroll
  for (int v = 0; v < 8; v++) {
    int o = o0 + v + half8;
    int oc = o & 255;
    int head = oc >> 6;
    int d = oc & 63;
    size_t hq = (((size_t)batch * HEADS + head) * NPIX + n) * DH + d;
    if (sec == 0)      qh[hq] = (_Float16)(acc[v] * QSCALE);
    else if (sec == 1) kh[hq] = (_Float16)acc[v];
    else vth[(((size_t)batch * HEADS + head) * DH + d) * NPIX + n] = (_Float16)acc[v];
  }
}

// -------------------------- relative-position bias -------------------------
// RW[bh,q=(x,y),jw] = q_scaled . rel_width[jw-y+W-1]
// RH[bh,q=(x,y),jh] = q_scaled . rel_height[jh-x+H-1]
__global__ void relbias_kernel(const _Float16* __restrict__ qh,
                               const float* __restrict__ rel_h,  // [2H-1,DH]
                               const float* __restrict__ rel_w,  // [2W-1,DH]
                               float* __restrict__ RW, float* __restrict__ RH) {
  int i = blockIdx.x * blockDim.x + threadIdx.x;
  if (i >= BDIM * HEADS * NPIX * WDIM) return;
  int j  = i & 63;
  int qn = (i >> 6) & (NPIX - 1);
  int bh = i >> 18;
  int x = qn >> 6, y = qn & 63;
  const _Float16* qp = qh + ((size_t)bh * NPIX + qn) * DH;
  const float* rw = rel_w + (size_t)(j - y + WDIM - 1) * DH;
  const float* rh = rel_h + (size_t)(j - x + HDIM - 1) * DH;
  float aw = 0.f, ah = 0.f;
#pragma unroll 8
  for (int d = 0; d < DH; d++) {
    float qv = (float)qp[d];
    aw += qv * rw[d];
    ah += qv * rh[d];
  }
  RW[i] = aw;
  RH[i] = ah;
}

// ---------------- async staging of one 64x64 f16 K tile + V^T tile ---------
// 128 threads each issue 4+4 global_load_async_to_lds_b128 (16B/lane);
// per-wave ASYNCcnt += 8. K tile is contiguous in global; V^T rows stride NPIX.
static __device__ __forceinline__ void stage_tile(unsigned ldsK, unsigned ldsV,
                                                  const _Float16* gk,
                                                  const _Float16* gv) {
  int tid = threadIdx.x;
#pragma unroll
  for (int it = 0; it < 4; ++it) {
    int e = (it * 128 + tid) * 8;          // half index within 64x64 tile
    asm volatile("global_load_async_to_lds_b128 %0, %1, off"
                 :: "v"(ldsK + (unsigned)e * 2u),
                    "v"((unsigned long long)(uintptr_t)(gk + e))
                 : "memory");
    int row = e >> 6, kc = e & 63;         // V^T: row = d, kc = key-in-tile
    asm volatile("global_load_async_to_lds_b128 %0, %1, off"
                 :: "v"(ldsV + (unsigned)e * 2u),
                    "v"((unsigned long long)(uintptr_t)(gv + (size_t)row * NPIX + kc))
                 : "memory");
  }
}

// ------------------------------ fused attention ----------------------------
// Flash-style: per wave a 16-row query tile; iterate 64-key tiles (== one jh
// row). K/V^T tiles double-buffered in LDS via async loads (shared by all 4
// waves), WMMA Q.K^T, add RW/RH bias, online softmax, WMMA P.V. Scores never
// touch HBM. grid (NPIX/64, HEADS, B), block 128.
__global__ void flash_kernel(const _Float16* __restrict__ qh,
                             const _Float16* __restrict__ kh,
                             const _Float16* __restrict__ vth,
                             const float* __restrict__ RW,
                             const float* __restrict__ RH,
                             _Float16* __restrict__ ao) {  // [B,N,C]
  __shared__ __align__(16) _Float16 ktile[2][64 * 64];  // [key][d]
  __shared__ __align__(16) _Float16 vtile[2][64 * 64];  // [d][key]
  __shared__ __align__(16) _Float16 plds[4][16 * 64];   // per-wave P staging

  int wave  = threadIdx.x >> 5;
  int lane  = threadIdx.x & 31;
  int col   = lane & 15;
  int half8 = (lane & 16) ? 8 : 0;
  int head = blockIdx.y, batch = blockIdx.z;
  int bh = batch * HEADS + head;
  int q0 = blockIdx.x * 64 + wave * 16;

  const _Float16* qp = qh + ((size_t)bh * NPIX + q0) * DH;
  v16h Aq0 = load_a(qp, DH, 0);
  v16h Aq1 = load_a(qp, DH, 32);

  float m[8], l[8];
  v8f O[4] = {};
#pragma unroll
  for (int v = 0; v < 8; v++) { m[v] = -3.0e38f; l[v] = 0.f; }

  const float* rwp = RW + ((size_t)bh * NPIX + q0) * 64;
  const float* rhp = RH + ((size_t)bh * NPIX + q0) * 64;

  const _Float16* kbase = kh + (size_t)bh * NPIX * DH;   // [key][d]
  const _Float16* vbase = vth + (size_t)bh * DH * NPIX;  // [d][key]
  unsigned ldsKa[2] = {(unsigned)(uintptr_t)&ktile[0][0],
                       (unsigned)(uintptr_t)&ktile[1][0]};
  unsigned ldsVa[2] = {(unsigned)(uintptr_t)&vtile[0][0],
                       (unsigned)(uintptr_t)&vtile[1][0]};

  stage_tile(ldsKa[0], ldsVa[0], kbase, vbase);  // prefetch jt = 0

  for (int jt = 0; jt < NPIX / 64; jt++) {
    int cur = jt & 1;
    if (jt + 1 < NPIX / 64) {
      // issue next tile into the other buffer, then wait only for current
      stage_tile(ldsKa[cur ^ 1], ldsVa[cur ^ 1],
                 kbase + (size_t)(jt + 1) * 64 * DH,
                 vbase + (size_t)(jt + 1) * 64);
      asm volatile("s_wait_asynccnt 0x8" ::: "memory");  // in-order: jt landed
    } else {
      asm volatile("s_wait_asynccnt 0x0" ::: "memory");
    }
    __syncthreads();  // all waves' portions of tile jt visible

    const _Float16* kp = &ktile[cur][0];
    const _Float16* vp = &vtile[cur][0];

    v8f S[4];
#pragma unroll
    for (int t = 0; t < 4; t++) {
      const _Float16* kb = kp + (size_t)t * 16 * 64;
      v8f s = {};
      s = wmma_f16(Aq0, load_b(kb, 64, 0),  s);
      s = wmma_f16(Aq1, load_b(kb, 64, 32), s);
      S[t] = s;
    }
    // bias: key tile jt covers jh == jt, jw = t*16+col
#pragma unroll
    for (int v = 0; v < 8; v++) {
      float rh = rhp[(size_t)(v + half8) * 64 + jt];
#pragma unroll
      for (int t = 0; t < 4; t++)
        S[t][v] += rh + rwp[(size_t)(v + half8) * 64 + t * 16 + col];
    }
    // online softmax (rows live across the 16 lanes of each half-wave)
#pragma unroll
    for (int v = 0; v < 8; v++) {
      float mx = S[0][v];
#pragma unroll
      for (int t = 1; t < 4; t++) mx = fmaxf(mx, S[t][v]);
#pragma unroll
      for (int d = 1; d < 16; d <<= 1) mx = fmaxf(mx, __shfl_xor(mx, d, 16));
      float mnew = fmaxf(m[v], mx);
      float corr = __expf(m[v] - mnew);
      float rsum = 0.f;
#pragma unroll
      for (int t = 0; t < 4; t++) {
        float p = __expf(S[t][v] - mnew);
        rsum += p;
        plds[wave][(v + half8) * 64 + t * 16 + col] = (_Float16)p;
      }
#pragma unroll
      for (int d = 1; d < 16; d <<= 1) rsum += __shfl_xor(rsum, d, 16);
      l[v] = l[v] * corr + rsum;
      m[v] = mnew;
#pragma unroll
      for (int t = 0; t < 4; t++) O[t][v] *= corr;
    }
    // P (C-layout in LDS) -> A-layout fragments, then P.V from LDS V^T rows
    v16h Pa0 = load_a(&plds[wave][0], 64, 0);
    v16h Pa1 = load_a(&plds[wave][0], 64, 32);
#pragma unroll
    for (int t = 0; t < 4; t++) {
      const _Float16* vb = vp + (size_t)t * 16 * 64;
      O[t] = wmma_f16(Pa0, load_b(vb, 64, 0),  O[t]);
      O[t] = wmma_f16(Pa1, load_b(vb, 64, 32), O[t]);
    }
    __syncthreads();  // reads done before next iteration overwrites buffer
  }
  // normalize and store ao[b][n][head*64 + d]
#pragma unroll
  for (int v = 0; v < 8; v++) {
    float inv = 1.0f / l[v];
#pragma unroll
    for (int t = 0; t < 4; t++) {
      size_t idx = ((size_t)batch * NPIX + q0 + v + half8) * CDIM +
                   (size_t)head * 64 + t * 16 + col;
      ao[idx] = (_Float16)(O[t][v] * inv);
    }
  }
}

// --------------------------- output projection + residual ------------------
__global__ void proj_kernel(const _Float16* __restrict__ wo,  // [C,C]
                            const _Float16* __restrict__ ao,  // [B,N,C]
                            const float* __restrict__ x,      // [B,C,N]
                            float* __restrict__ out) {        // [B,C,N]
  int wave = threadIdx.x >> 5;
  int lane = threadIdx.x & 31;
  int n0 = (blockIdx.x * 4 + wave) * 16;
  int o0 = blockIdx.y * 16;
  int batch = blockIdx.z;
  const _Float16* a0 = wo + (size_t)o0 * CDIM;
  const _Float16* b0 = ao + ((size_t)batch * NPIX + n0) * CDIM;
  v8f acc = {};
#pragma unroll
  for (int kk = 0; kk < CDIM; kk += 32)
    acc = wmma_f16(load_a(a0, CDIM, kk), load_b(b0, CDIM, kk), acc);

  int col   = lane & 15;
  int half8 = (lane & 16) ? 8 : 0;
  int n = n0 + col;
#pragma unroll
  for (int v = 0; v < 8; v++) {
    int o = o0 + v + half8;
    size_t idx = ((size_t)batch * CDIM + o) * NPIX + n;
    out[idx] = acc[v] + x[idx];
  }
}

// ------------------------------- launcher ----------------------------------
extern "C" void kernel_launch(void* const* d_in, const int* in_sizes, int n_in,
                              void* d_out, int out_size, void* d_ws, size_t ws_size,
                              hipStream_t stream) {
  (void)in_sizes; (void)n_in; (void)out_size; (void)ws_size;
  const float* x     = (const float*)d_in[0];
  const float* w_qkv = (const float*)d_in[1];
  const float* w_out = (const float*)d_in[2];
  const float* rel_h = (const float*)d_in[3];
  const float* rel_w = (const float*)d_in[4];
  float* out = (float*)d_out;

  // workspace sub-allocation (~37 MB, all 16B aligned)
  char* ws = (char*)d_ws;
  size_t off = 0;
  _Float16* xt  = (_Float16*)(ws + off); off += (size_t)BDIM * NPIX * CDIM * 2;
  _Float16* qh  = (_Float16*)(ws + off); off += (size_t)BDIM * HEADS * NPIX * DH * 2;
  _Float16* kh  = (_Float16*)(ws + off); off += (size_t)BDIM * HEADS * NPIX * DH * 2;
  _Float16* vth = (_Float16*)(ws + off); off += (size_t)BDIM * HEADS * DH * NPIX * 2;
  float*    RW  = (float*)(ws + off);    off += (size_t)BDIM * HEADS * NPIX * 64 * 4;
  float*    RH  = (float*)(ws + off);    off += (size_t)BDIM * HEADS * NPIX * 64 * 4;
  _Float16* ao  = (_Float16*)(ws + off); off += (size_t)BDIM * NPIX * CDIM * 2;
  _Float16* wq16 = (_Float16*)(ws + off); off += (size_t)3 * CDIM * CDIM * 2;
  _Float16* wo16 = (_Float16*)(ws + off); off += (size_t)CDIM * CDIM * 2;

  cvt_xt_kernel<<<(BDIM * CDIM * NPIX + 255) / 256, 256, 0, stream>>>(x, xt);
  cvt_f16_kernel<<<(3 * CDIM * CDIM + 255) / 256, 256, 0, stream>>>(w_qkv, wq16,
                                                                    3 * CDIM * CDIM);
  cvt_f16_kernel<<<(CDIM * CDIM + 255) / 256, 256, 0, stream>>>(w_out, wo16,
                                                                CDIM * CDIM);
  qkv_kernel<<<dim3(NPIX / 64, 3 * CDIM / 16, BDIM), 128, 0, stream>>>(
      wq16, xt, qh, kh, vth);
  relbias_kernel<<<(BDIM * HEADS * NPIX * WDIM + 255) / 256, 256, 0, stream>>>(
      qh, rel_h, rel_w, RW, RH);
  flash_kernel<<<dim3(NPIX / 64, HEADS, BDIM), 128, 0, stream>>>(
      qh, kh, vth, RW, RH, ao);
  proj_kernel<<<dim3(NPIX / 64, CDIM / 16, BDIM), 128, 0, stream>>>(
      wo16, ao, x, out);
}
